// Head_63273458204707
// MI455X (gfx1250) — matrix operand order repaired and checked by reading, not compiled, not run on hardware
//
#include <hip/hip_runtime.h>

// Attention head: x[8192,2048] -> q,k,v[8192,128] -> softmax(qk^T/sqrt(128))v
// Stage 1: 3x GEMM (bf16 WMMA, f32 accum), Q/K row-major bf16, V transposed bf16.
// Stage 2: flash attention; K/V chunks staged to LDS with gfx1250 async
//          global->LDS loads (double buffered, s_wait_asynccnt), all matmuls
//          via v_wmma_f32_16x16x32_bf16.

#define N_TOK 8192
#define D_DIM 2048
#define H_DIM 128

typedef __attribute__((ext_vector_type(16))) __bf16 v16bf;
typedef __attribute__((ext_vector_type(8)))  float  v8f;
typedef int v4i_vs __attribute__((vector_size(16)));   // matches async builtin param

union V16U { v16bf v; uint4 u[2]; };

#if __has_builtin(__builtin_amdgcn_global_load_async_to_lds_b128) && \
    __has_builtin(__builtin_amdgcn_s_wait_asynccnt)
#define HAVE_ASYNC 1
#endif

// 16-byte global -> LDS copy. Async (ASYNCcnt-tracked) when available.
__device__ inline void copy16_g2l(void* lds, const void* gmem) {
#ifdef HAVE_ASYNC
    __builtin_amdgcn_global_load_async_to_lds_b128(
        (__attribute__((address_space(1))) v4i_vs*)gmem,
        (__attribute__((address_space(3))) v4i_vs*)lds,
        /*offset=*/0, /*cpol=*/0);
#else
    *(uint4*)lds = *(const uint4*)gmem;
#endif
}

__device__ inline void wait_async_le(int n) {
#ifdef HAVE_ASYNC
    if (n == 0) __builtin_amdgcn_s_wait_asynccnt(0);
    else        __builtin_amdgcn_s_wait_asynccnt(8);
#endif
}

// Native fptrunc (RNE); lets the backend use packed bf16 cvt instructions.
__device__ inline __bf16 f2bf(float f) { return (__bf16)f; }

__device__ inline float fast_rcp(float x) {
#if __has_builtin(__builtin_amdgcn_rcpf)
    return __builtin_amdgcn_rcpf(x);
#else
    return 1.0f / x;
#endif
}

__device__ inline v8f wmma_bf16(v16bf a, v16bf b, v8f c) {
    return __builtin_amdgcn_wmma_f32_16x16x32_bf16(
        /*neg_a=*/false, a, /*neg_b=*/false, b,
        /*c_mod=*/(short)0, c, /*reuse_a=*/false, /*reuse_b=*/false);
}

// A fragment (16x32, MxK) from bf16 row-major: element (m,k) at base[m*ld+k].
// Lane layout: m = lane&15; e<8 -> k = 8g+e ; e>=8 -> k = 16+8g+(e-8), g=lane>>4.
__device__ inline v16bf load_a(const __bf16* base, int ld, int lane) {
    int g = lane >> 4, m = lane & 15;
    const __bf16* p0 = base + m * ld + 8 * g;
    const __bf16* p1 = base + m * ld + 16 + 8 * g;
    V16U r;
    r.u[0] = *(const uint4*)p0;
    r.u[1] = *(const uint4*)p1;
    return r.v;
}

// B fragment (32x16, KxN): element (k,n) at base[n*ldn + k].
// Lane layout: n = lane&15; e<8 -> k = 8g+e ; e>=8 -> k = 16+8g+(e-8), g=lane>>4.
__device__ inline v16bf load_b(const __bf16* base, int ldn, int lane) {
    int g = lane >> 4, n = lane & 15;
    const __bf16* p = base + n * ldn + 8 * g;
    V16U r;
    r.u[0] = *(const uint4*)p;
    r.u[1] = *(const uint4*)(p + 16);
    return r.v;
}

// A fragment from fp32 row-major with convert-to-bf16.
__device__ inline v16bf load_a_f32(const float* base, int ld, int lane) {
    int g = lane >> 4, m = lane & 15;
    const float4* p0 = (const float4*)(base + m * ld + 8 * g);
    const float4* p1 = (const float4*)(base + m * ld + 16 + 8 * g);
    float4 f0 = p0[0], f1 = p0[1], f2 = p1[0], f3 = p1[1];
    v16bf a;
    a[0]  = f2bf(f0.x); a[1]  = f2bf(f0.y); a[2]  = f2bf(f0.z); a[3]  = f2bf(f0.w);
    a[4]  = f2bf(f1.x); a[5]  = f2bf(f1.y); a[6]  = f2bf(f1.z); a[7]  = f2bf(f1.w);
    a[8]  = f2bf(f2.x); a[9]  = f2bf(f2.y); a[10] = f2bf(f2.z); a[11] = f2bf(f2.w);
    a[12] = f2bf(f3.x); a[13] = f2bf(f3.y); a[14] = f2bf(f3.z); a[15] = f2bf(f3.w);
    return a;
}

// ---------------- Stage 1: out = x @ W + b  (bf16 output) ----------------
// Block: 256 threads (8 waves), 128 rows x 128 cols per block.
// Wave w: rows (w&3)*32 (two 16-row tiles), cols (w>>2)*64 (four 16-col tiles).
__global__ __launch_bounds__(256) void qkv_proj(
        const float* __restrict__ x, const float* __restrict__ W,
        const float* __restrict__ bias, __bf16* __restrict__ out,
        int transposed) {
    __shared__ __bf16 ldsW[H_DIM * 32];   // Wt[h][d] for current 32-deep slice

    const int tid  = threadIdx.x;
    const int lane = tid & 31;
    const int w    = tid >> 5;
    const int rowBase = blockIdx.x * 128 + (w & 3) * 32;
    const int colBase = (w >> 2) * 64;

    const v8f z8 = {0.f, 0.f, 0.f, 0.f, 0.f, 0.f, 0.f, 0.f};
    v8f acc[2][4];
#pragma unroll
    for (int r = 0; r < 2; ++r)
#pragma unroll
        for (int c = 0; c < 4; ++c) acc[r][c] = z8;

    for (int d0 = 0; d0 < D_DIM; d0 += 32) {
        // Stage W[d0:d0+32, 0:128] transposed into LDS as bf16.
#pragma unroll
        for (int i = 0; i < 4; ++i) {
            int li = tid + i * 256;           // 0..1023 float4s
            int d  = li >> 5;                 // 0..31
            int h4 = (li & 31) << 2;          // 0,4,...,124
            float4 wv = *(const float4*)(W + (size_t)(d0 + d) * H_DIM + h4);
            ldsW[(h4 + 0) * 32 + d] = f2bf(wv.x);
            ldsW[(h4 + 1) * 32 + d] = f2bf(wv.y);
            ldsW[(h4 + 2) * 32 + d] = f2bf(wv.z);
            ldsW[(h4 + 3) * 32 + d] = f2bf(wv.w);
        }
        __syncthreads();

        v16bf a0 = load_a_f32(x + (size_t)rowBase * D_DIM + d0, D_DIM, lane);
        v16bf a1 = load_a_f32(x + (size_t)(rowBase + 16) * D_DIM + d0, D_DIM, lane);
#pragma unroll
        for (int c = 0; c < 4; ++c) {
            v16bf bfrag = load_b(ldsW + (colBase + c * 16) * 32, 32, lane);
            acc[0][c] = wmma_bf16(a0, bfrag, acc[0][c]);
            acc[1][c] = wmma_bf16(a1, bfrag, acc[1][c]);
        }
        __syncthreads();
    }

    const int g = lane >> 4, n0 = lane & 15;
    if (transposed) {
#pragma unroll
        for (int r = 0; r < 2; ++r)
#pragma unroll
            for (int c = 0; c < 4; ++c) {
                int h = colBase + c * 16 + n0;
                float bv = bias[h];
#pragma unroll
                for (int i = 0; i < 8; ++i) {
                    int row = rowBase + r * 16 + i + 8 * g;
                    out[(size_t)h * N_TOK + row] = f2bf(acc[r][c][i] + bv);
                }
            }
    } else {
#pragma unroll
        for (int r = 0; r < 2; ++r)
#pragma unroll
            for (int c = 0; c < 4; ++c) {
                int h = colBase + c * 16 + n0;
                float bv = bias[h];
#pragma unroll
                for (int i = 0; i < 8; ++i) {
                    int row = rowBase + r * 16 + i + 8 * g;
                    out[(size_t)row * H_DIM + h] = f2bf(acc[r][c][i] + bv);
                }
            }
    }
}

// ---------------- Stage 2: flash attention ----------------
// Block: 128 threads (4 waves), each wave owns 16 query rows.
// Keys processed in chunks of 32. K chunk (32x128) and V chunk (128x32, from
// V^T) staged into LDS via async global->LDS loads, double buffered.
__global__ __launch_bounds__(128) void attn(
        const __bf16* __restrict__ Q, const __bf16* __restrict__ K,
        const __bf16* __restrict__ Vt, float* __restrict__ Out) {
    __shared__ __bf16 ldsK[2][32 * H_DIM];   // [key][h] chunk, 2 x 8KB
    __shared__ __bf16 ldsV[2][H_DIM * 32];   // [h][key] chunk, 2 x 8KB
    __shared__ __bf16 ldsP[4][16 * 32];      // per-wave P tile

    const int tid   = threadIdx.x;
    const int lane  = tid & 31;
    const int w     = tid >> 5;
    const int qBase = blockIdx.x * 64 + w * 16;
    const int g = lane >> 4, n0 = lane & 15;
    const float scale = 0.08838834764831845f;   // 1/sqrt(128)

    // Stage one 32-key chunk: 8 async b128 per thread (4 K + 4 V).
    auto stage = [&](int key0, int buf) {
#pragma unroll
        for (int j = 0; j < 4; ++j) {
            int c = tid + j * 128;            // 0..511 16B-chunks
            // K rows key0..key0+31 are 8KB contiguous.
            copy16_g2l((char*)&ldsK[buf][0] + c * 16,
                       (const char*)K + (size_t)key0 * H_DIM * 2 + c * 16);
            // V^T rows: 64B per h row, strided by N in global.
            int h = c >> 2, off = (c & 3) * 8;
            copy16_g2l((char*)&ldsV[buf][0] + h * 64 + off * 2,
                       (const char*)Vt + ((size_t)h * N_TOK + key0 + off) * 2);
        }
    };

    // Resident Q fragments covering H=128 (4 k-steps of 32).
    v16bf qf[4];
#pragma unroll
    for (int t = 0; t < 4; ++t)
        qf[t] = load_a(Q + (size_t)qBase * H_DIM + t * 32, H_DIM, lane);

    const v8f z8 = {0.f, 0.f, 0.f, 0.f, 0.f, 0.f, 0.f, 0.f};
    v8f acc[8];
    float m_st[8], l_st[8];
#pragma unroll
    for (int j = 0; j < 8; ++j) acc[j] = z8;
#pragma unroll
    for (int i = 0; i < 8; ++i) { m_st[i] = -1e30f; l_st[i] = 0.f; }

    __bf16* myP = &ldsP[w][0];

    stage(0, 0);

    for (int key0 = 0; key0 < N_TOK; key0 += 32) {
        const int buf = (key0 >> 5) & 1;
        if (key0 + 32 < N_TOK) {
            stage(key0 + 32, buf ^ 1);  // prefetch next chunk
            wait_async_le(8);           // current chunk's 8 loads done
        } else {
            wait_async_le(0);
        }
        __syncthreads();

        // S = Q . K^T for 16 rows x 32 keys (two 16x16 tiles).
        v8f s0 = z8, s1 = z8;
#pragma unroll
        for (int t = 0; t < 4; ++t) {
            v16bf kb0 = load_b(&ldsK[buf][t * 32], H_DIM, lane);
            v16bf kb1 = load_b(&ldsK[buf][16 * H_DIM + t * 32], H_DIM, lane);
            s0 = wmma_bf16(qf[t], kb0, s0);
            s1 = wmma_bf16(qf[t], kb1, s1);
        }

        // Online softmax; row i of the C fragment = i + 8*(lane>=16), columns
        // across the 16-lane half, so xor masks 1,2,4,8 reduce+broadcast.
        float alpha[8];
#pragma unroll
        for (int i = 0; i < 8; ++i) {
            float a = s0[i] * scale, b = s1[i] * scale;
            float mx = fmaxf(a, b);
            mx = fmaxf(mx, __shfl_xor(mx, 1, 32));
            mx = fmaxf(mx, __shfl_xor(mx, 2, 32));
            mx = fmaxf(mx, __shfl_xor(mx, 4, 32));
            mx = fmaxf(mx, __shfl_xor(mx, 8, 32));
            float m_new = fmaxf(m_st[i], mx);
            alpha[i] = __expf(m_st[i] - m_new);
            m_st[i] = m_new;
            float p0 = __expf(a - m_new);
            float p1 = __expf(b - m_new);
            float rs = p0 + p1;
            rs += __shfl_xor(rs, 1, 32);
            rs += __shfl_xor(rs, 2, 32);
            rs += __shfl_xor(rs, 4, 32);
            rs += __shfl_xor(rs, 8, 32);
            l_st[i] = l_st[i] * alpha[i] + rs;
            // C-layout -> LDS row-major 16x32 (A-fragment reload; same-wave DS
            // ops are in-order, compiler inserts s_wait_dscnt).
            myP[(i + 8 * g) * 32 + n0]      = f2bf(p0);
            myP[(i + 8 * g) * 32 + n0 + 16] = f2bf(p1);
        }

#pragma unroll
        for (int j = 0; j < 8; ++j)
#pragma unroll
            for (int i = 0; i < 8; ++i) acc[j][i] *= alpha[i];

        v16bf pf = load_a(myP, 32, lane);
#pragma unroll
        for (int j = 0; j < 8; ++j) {
            v16bf vb = load_b(&ldsV[buf][(j * 16) * 32], 32, lane);
            acc[j] = wmma_bf16(pf, vb, acc[j]);
        }
        __syncthreads();
    }

#pragma unroll
    for (int i = 0; i < 8; ++i) l_st[i] = fast_rcp(l_st[i]);
#pragma unroll
    for (int j = 0; j < 8; ++j) {
        int h = j * 16 + n0;
#pragma unroll
        for (int i = 0; i < 8; ++i) {
            int row = qBase + i + 8 * g;
            Out[(size_t)row * H_DIM + h] = acc[j][i] * l_st[i];
        }
    }
}

extern "C" void kernel_launch(void* const* d_in, const int* in_sizes, int n_in,
                              void* d_out, int out_size, void* d_ws, size_t ws_size,
                              hipStream_t stream) {
    // setup_inputs order: x, Wk, bk, Wq, bq, Wv, bv
    const float* x  = (const float*)d_in[0];
    const float* Wk = (const float*)d_in[1];
    const float* bk = (const float*)d_in[2];
    const float* Wq = (const float*)d_in[3];
    const float* bq = (const float*)d_in[4];
    const float* Wv = (const float*)d_in[5];
    const float* bv = (const float*)d_in[6];
    float* out = (float*)d_out;

    __bf16* Qw = (__bf16*)d_ws;                      // [N,H] bf16, 2 MB
    __bf16* Kw = Qw + (size_t)N_TOK * H_DIM;         // [N,H] bf16, 2 MB
    __bf16* Vt = Kw + (size_t)N_TOK * H_DIM;         // [H,N] bf16, 2 MB

    dim3 blk1(256), grd1(N_TOK / 128);
    qkv_proj<<<grd1, blk1, 0, stream>>>(x, Wq, bq, Qw, 0);
    qkv_proj<<<grd1, blk1, 0, stream>>>(x, Wk, bk, Kw, 0);
    qkv_proj<<<grd1, blk1, 0, stream>>>(x, Wv, bv, Vt, 1);

    attn<<<dim3(N_TOK / 64), dim3(128), 0, stream>>>(Qw, Kw, Vt, out);
}
